// selfAttention_50749333570077
// MI455X (gfx1250) — matrix operand order repaired and checked by reading, not compiled, not run on hardware
//
#include <hip/hip_runtime.h>

// Self-attention  B=4, S=4096, D=512; reference scale = B/2 = 2.0 (not sqrt).
// All GEMMs on v_wmma_f32_16x16x32_bf16, f32 accumulation.
// - prep kernels convert x/W to bf16 once (hardware cvt)
// - projection GEMM: pure-register pipeline out of the 192MB L2 (no LDS)
// - attention: 32 queries/block, K and V tiles double-buffered in LDS via
//   global_load_async_to_lds_b128 (ASYNCcnt), wave-parallel online softmax.
// Needs ~69 MB of d_ws.

#define DHID  512
#define BATCH 4
#define SEQ   4096
#define NKT   (SEQ / 32)   // key tiles of 32

typedef __bf16 bf16_t;
typedef __attribute__((ext_vector_type(16))) __bf16 bf16x16;
typedef __attribute__((ext_vector_type(8)))  __bf16 bf16x8;
typedef __attribute__((ext_vector_type(4)))  __bf16 bf16x4;
typedef __attribute__((ext_vector_type(8)))  float  f32x8;
typedef __attribute__((ext_vector_type(4)))  float  f32x4;

union Frag  { bf16x16 v; bf16x8 h[2]; };
union Pack8 { bf16x8  v; bf16x4 q[2]; };

#if __has_builtin(__builtin_amdgcn_global_load_async_to_lds_b128)
#define USE_ASYNC_LDS 1
#else
#define USE_ASYNC_LDS 0
#endif

#if USE_ASYNC_LDS
// Exact parameter types per hipcc diagnostic: GCC-vector int4 pointers in
// global (addrspace 1) and shared (addrspace 3) address spaces.
typedef int v4i_ __attribute__((vector_size(16)));
typedef __attribute__((address_space(1))) v4i_ gv4i;
typedef __attribute__((address_space(3))) v4i_ lv4i;
#endif

// Copy 16 bytes global -> LDS.  Async (ASYNCcnt-tracked) when available.
__device__ __forceinline__ void copy16_g2l(const bf16_t* g, bf16_t* l) {
#if USE_ASYNC_LDS
    gv4i* gp = (gv4i*)(unsigned long long)(const void*)g;
    lv4i* lp = (lv4i*)(unsigned int)(unsigned long long)(void*)l;
    __builtin_amdgcn_global_load_async_to_lds_b128(gp, lp, 0, 0);
#else
    *(bf16x8*)l = *(const bf16x8*)g;
#endif
}

__device__ __forceinline__ void wait_async0() {
#if USE_ASYNC_LDS
#if __has_builtin(__builtin_amdgcn_s_wait_asynccnt)
    __builtin_amdgcn_s_wait_asynccnt(0);
#else
    asm volatile("s_wait_asynccnt 0x0" ::: "memory");
#endif
#endif
}

// ---------------------------------------------------------------------------
// Prep 1: x (f32) -> xb (bf16), hardware cvt, 8 elements/thread.
// ---------------------------------------------------------------------------
__global__ __launch_bounds__(256)
void cvt_x_kernel(const float* __restrict__ x, bf16_t* __restrict__ xb)
{
    const size_t i = ((size_t)blockIdx.x * 256 + threadIdx.x) * 8;
    f32x4 a = *(const f32x4*)(x + i);
    f32x4 b = *(const f32x4*)(x + i + 4);
    Pack8 p;
    p.q[0] = __builtin_convertvector(a, bf16x4);
    p.q[1] = __builtin_convertvector(b, bf16x4);
    *(bf16x8*)(xb + i) = p.v;
}

// ---------------------------------------------------------------------------
// Prep 2: W[k][n] (f32) -> Wt[z][n][k] (bf16, transposed).
// ---------------------------------------------------------------------------
__global__ __launch_bounds__(256)
void cvt_w_kernel(const float* __restrict__ Wq, const float* __restrict__ Wk,
                  const float* __restrict__ Wv, bf16_t* __restrict__ WtB)
{
    const int z = blockIdx.z;
    const float* W  = (z == 0) ? Wq : (z == 1) ? Wk : Wv;
    bf16_t*     Wt  = WtB + (size_t)z * DHID * DHID;
    const int idx = blockIdx.x * 256 + threadIdx.x;   // 0..32767
    const int k   = idx >> 6;                          // 0..511
    const int nb  = (idx & 63) * 8;
    f32x4 a = *(const f32x4*)(W + (size_t)k * DHID + nb);
    f32x4 b = *(const f32x4*)(W + (size_t)k * DHID + nb + 4);
    bf16x4 c0 = __builtin_convertvector(a, bf16x4);
    bf16x4 c1 = __builtin_convertvector(b, bf16x4);
#pragma unroll
    for (int i = 0; i < 4; ++i) {
        Wt[(size_t)(nb + i)     * DHID + k] = c0[i];
        Wt[(size_t)(nb + 4 + i) * DHID + k] = c1[i];
    }
}

// ---------------------------------------------------------------------------
// Kernel 1: Q/K = xb @ W + b (row-major bf16); V stored TRANSPOSED
// ([batch][dim][key]).  Pure register GEMM out of L2: no LDS, no barriers.
// ---------------------------------------------------------------------------
__global__ __launch_bounds__(256)
void qkv_kernel(const bf16_t* __restrict__ xb, const bf16_t* __restrict__ WtB,
                const float* __restrict__ bq, const float* __restrict__ bk,
                const float* __restrict__ bv,
                bf16_t* __restrict__ Qb, bf16_t* __restrict__ Kb,
                bf16_t* __restrict__ VT)
{
    const int z = blockIdx.z;
    const bf16_t* W    = WtB + (size_t)z * DHID * DHID;   // [n][k] bf16
    const float*  bias = (z == 0) ? bq : (z == 1) ? bk : bv;

    const int m0   = blockIdx.x * 64;
    const int n0   = blockIdx.y * 64;
    const int lane = threadIdx.x & 31;
    const int w    = threadIdx.x >> 5;
    const int half = lane >> 4;
    const int l16  = lane & 15;
    const int mW   = w & 3;
    const int nH   = w >> 2;

    const int    mrow  = m0 + mW * 16 + l16;
    const int    ncol0 = n0 + nH * 32 + l16;
    const int    ncol1 = ncol0 + 16;
    const bf16_t* arow = xb + (size_t)mrow * DHID;
    const bf16_t* b0r  = W + (size_t)ncol0 * DHID;
    const bf16_t* b1r  = W + (size_t)ncol1 * DHID;

    f32x8 acc0 = {};
    f32x8 acc1 = {};

#pragma unroll 4
    for (int k0 = 0; k0 < DHID; k0 += 32) {
        Frag a, b0, b1;
        a.h[0]  = *(const bf16x8*)(arow + k0 + half * 8);
        a.h[1]  = *(const bf16x8*)(arow + k0 + half * 8 + 16);
        b0.h[0] = *(const bf16x8*)(b0r + k0 + half * 8);
        b0.h[1] = *(const bf16x8*)(b0r + k0 + half * 8 + 16);
        b1.h[0] = *(const bf16x8*)(b1r + k0 + half * 8);
        b1.h[1] = *(const bf16x8*)(b1r + k0 + half * 8 + 16);
        acc0 = __builtin_amdgcn_wmma_f32_16x16x32_bf16(
                   false, a.v, false, b0.v, (short)0, acc0, false, false);
        acc1 = __builtin_amdgcn_wmma_f32_16x16x32_bf16(
                   false, a.v, false, b1.v, (short)0, acc1, false, false);
    }

    const float bia0 = bias[ncol0];
    const float bia1 = bias[ncol1];
#pragma unroll
    for (int v = 0; v < 8; ++v) {
        const int row = m0 + mW * 16 + v + half * 8;
        const bf16_t o0 = (bf16_t)(acc0[v] + bia0);
        const bf16_t o1 = (bf16_t)(acc1[v] + bia1);
        if (z == 0) {
            Qb[(size_t)row * DHID + ncol0] = o0;
            Qb[(size_t)row * DHID + ncol1] = o1;
        } else if (z == 1) {
            Kb[(size_t)row * DHID + ncol0] = o0;
            Kb[(size_t)row * DHID + ncol1] = o1;
        } else {
            const int bb = row >> 12;       // batch
            const int ss = row & (SEQ - 1); // key index
            VT[((size_t)bb * DHID + ncol0) * SEQ + ss] = o0;
            VT[((size_t)bb * DHID + ncol1) * SEQ + ss] = o1;
        }
    }
}

// ---------------------------------------------------------------------------
// Kernel 2: flash attention.  Block = (32 queries, one batch), 8 waves each
// owning a 64-wide D chunk.  K AND V tiles double-buffered in LDS (async);
// each staged tile serves two 16-query passes; V B-fragments reused in regs.
// ---------------------------------------------------------------------------
__global__ __launch_bounds__(256)
void attn_kernel(const bf16_t* __restrict__ Qb, const bf16_t* __restrict__ Kb,
                 const bf16_t* __restrict__ VT, float* __restrict__ out)
{
    const int q0   = blockIdx.x * 32;
    const int b    = blockIdx.y;
    const int tid  = threadIdx.x;
    const int lane = tid & 31;
    const int w    = tid >> 5;
    const int half = lane >> 4;
    const int l16  = lane & 15;
    const size_t rbase = (size_t)b * SEQ;

    __shared__ __align__(16) bf16_t Kt[2][32][DHID];   // K tile, double-buffered
    __shared__ __align__(16) bf16_t Vt[2][DHID][32];   // V tile (dim-major)
    __shared__ __align__(16) float  Sc[32][32];        // score tiles (2 passes)
    __shared__ __align__(16) bf16_t P[32][32];         // probabilities
    __shared__ __align__(16) float  mrow[32], lrow[32], frow[32];

    if (tid < 32) { mrow[tid] = -1e30f; lrow[tid] = 0.0f; }

    f32x8 zacc[2][4] = {};

    // Hoisted Q A-fragments: 2 query tiles x 2 K-steps over this wave's chunk.
    Frag qa[2][2];
#pragma unroll
    for (int qi = 0; qi < 2; ++qi) {
        const bf16_t* qr = Qb + (rbase + q0 + qi * 16 + l16) * DHID
                              + w * 64 + half * 8;
#pragma unroll
        for (int s = 0; s < 2; ++s) {
            qa[qi][s].h[0] = *(const bf16x8*)(qr + s * 32);
            qa[qi][s].h[1] = *(const bf16x8*)(qr + s * 32 + 16);
        }
    }

    // Staging assignments (8 x 16B each for K and for V per thread).
    const int krow = tid >> 3;            // K: key row 0..31
    const int kcol = (tid & 7) * 64;      // K: element col base (128B)
    const int vd0  = tid * 2;             // V: two dim rows (64B each)

    // Prime buffer 0 with tile 0.
    {
        const bf16_t* ks = Kb + (rbase + krow) * DHID + kcol;
#pragma unroll
        for (int j = 0; j < 8; ++j)
            copy16_g2l(ks + j * 8, &Kt[0][krow][kcol + j * 8]);
#pragma unroll
        for (int d = 0; d < 2; ++d) {
            const bf16_t* vs = VT + ((size_t)b * DHID + vd0 + d) * SEQ;
#pragma unroll
            for (int j = 0; j < 4; ++j)
                copy16_g2l(vs + j * 8, &Vt[0][vd0 + d][j * 8]);
        }
    }

    for (int kt = 0; kt < NKT; ++kt) {
        const int k0  = kt * 32;
        const int cur = kt & 1;

        // 1) zero score tiles (32x32 f32, one b128 per thread)
        ((f32x4*)Sc)[tid] = (f32x4){0.f, 0.f, 0.f, 0.f};

        // 2) current tiles resident + all waves synced
        wait_async0();
        __syncthreads();

        // 3) issue async copies of next K/V tiles into the other buffer
        if (kt + 1 < NKT) {
            const bf16_t* ks = Kb + (rbase + k0 + 32 + krow) * DHID + kcol;
#pragma unroll
            for (int j = 0; j < 8; ++j)
                copy16_g2l(ks + j * 8, &Kt[1 - cur][krow][kcol + j * 8]);
#pragma unroll
            for (int d = 0; d < 2; ++d) {
                const bf16_t* vs = VT + ((size_t)b * DHID + vd0 + d) * SEQ
                                      + k0 + 32;
#pragma unroll
                for (int j = 0; j < 4; ++j)
                    copy16_g2l(vs + j * 8, &Vt[1 - cur][vd0 + d][j * 8]);
            }
        }

        // 4) partial scores for both query tiles (Kt reused)
#pragma unroll
        for (int qi = 0; qi < 2; ++qi) {
#pragma unroll
            for (int nt = 0; nt < 2; ++nt) {
                f32x8 s = {};
#pragma unroll
                for (int ks = 0; ks < 2; ++ks) {
                    Frag kf;
                    const bf16_t* kr = &Kt[cur][nt * 16 + l16]
                                          [w * 64 + ks * 32 + half * 8];
                    kf.h[0] = *(const bf16x8*)(kr);
                    kf.h[1] = *(const bf16x8*)(kr + 16);
                    s = __builtin_amdgcn_wmma_f32_16x16x32_bf16(
                            false, qa[qi][ks].v, false, kf.v, (short)0, s,
                            false, false);
                }
#pragma unroll
                for (int v = 0; v < 8; ++v)
                    atomicAdd(&Sc[qi * 16 + v + half * 8][nt * 16 + l16], s[v]);
            }
        }
        __syncthreads();

        // 5) wave-parallel online softmax: 8-lane group per row, 4 cols/lane
        {
            const int grp = lane >> 3;            // 0..3
            const int gl  = lane & 7;             // lane in group
            const int row = w * 4 + grp;          // 0..31
            const int c0  = gl * 4;
            f32x4 sv = *(const f32x4*)&Sc[row][c0];
            float s0 = sv[0] * 0.5f, s1 = sv[1] * 0.5f;   // scale = 1/2.0
            float s2 = sv[2] * 0.5f, s3 = sv[3] * 0.5f;
            float mx = fmaxf(fmaxf(s0, s1), fmaxf(s2, s3));
            mx = fmaxf(mx, __shfl_xor(mx, 1, 32));
            mx = fmaxf(mx, __shfl_xor(mx, 2, 32));
            mx = fmaxf(mx, __shfl_xor(mx, 4, 32));
            const float m_old = mrow[row];
            const float mnew  = fmaxf(m_old, mx);
            float p0 = __expf(s0 - mnew), p1 = __expf(s1 - mnew);
            float p2 = __expf(s2 - mnew), p3 = __expf(s3 - mnew);
            bf16x4 pb;
            pb[0] = (bf16_t)p0; pb[1] = (bf16_t)p1;
            pb[2] = (bf16_t)p2; pb[3] = (bf16_t)p3;
            *(bf16x4*)&P[row][c0] = pb;
            float sum = (p0 + p1) + (p2 + p3);
            sum += __shfl_xor(sum, 1, 32);
            sum += __shfl_xor(sum, 2, 32);
            sum += __shfl_xor(sum, 4, 32);
            if (gl == 0) {
                const float f = __expf(m_old - mnew);
                mrow[row] = mnew;
                frow[row] = f;
                lrow[row] = lrow[row] * f + sum;
            }
        }
        __syncthreads();

        // 6+7) rescale Z and accumulate P @ V (V fragments shared by passes)
        {
            Frag vf[4];
#pragma unroll
            for (int nt = 0; nt < 4; ++nt) {
                const bf16_t* vp = &Vt[cur][w * 64 + nt * 16 + l16][half * 8];
                vf[nt].h[0] = *(const bf16x8*)(vp);
                vf[nt].h[1] = *(const bf16x8*)(vp + 16);
            }
#pragma unroll
            for (int qi = 0; qi < 2; ++qi) {
                f32x4 f0 = *(const f32x4*)&frow[qi * 16 + half * 8];
                f32x4 f1 = *(const f32x4*)&frow[qi * 16 + half * 8 + 4];
#pragma unroll
                for (int nt = 0; nt < 4; ++nt)
#pragma unroll
                    for (int v = 0; v < 4; ++v) {
                        zacc[qi][nt][v]     *= f0[v];
                        zacc[qi][nt][v + 4] *= f1[v];
                    }
                Frag pa;
                const bf16_t* pp = &P[qi * 16 + l16][half * 8];
                pa.h[0] = *(const bf16x8*)(pp);
                pa.h[1] = *(const bf16x8*)(pp + 16);
#pragma unroll
                for (int nt = 0; nt < 4; ++nt)
                    zacc[qi][nt] = __builtin_amdgcn_wmma_f32_16x16x32_bf16(
                                       false, pa.v, false, vf[nt].v, (short)0,
                                       zacc[qi][nt], false, false);
            }
        }
        // next iteration's barrier orders buffer reuse
    }

    // finalize: Z / l -> f32 output
#pragma unroll
    for (int qi = 0; qi < 2; ++qi) {
        f32x4 l0 = *(const f32x4*)&lrow[qi * 16 + half * 8];
        f32x4 l1 = *(const f32x4*)&lrow[qi * 16 + half * 8 + 4];
        float inv[8];
#pragma unroll
        for (int i = 0; i < 4; ++i) {
            inv[i]     = 1.0f / l0[i];
            inv[i + 4] = 1.0f / l1[i];
        }
#pragma unroll
        for (int nt = 0; nt < 4; ++nt)
#pragma unroll
            for (int v = 0; v < 8; ++v) {
                const size_t row = rbase + q0 + qi * 16 + v + half * 8;
                out[row * DHID + w * 64 + nt * 16 + l16] =
                    zacc[qi][nt][v] * inv[v];
            }
    }
}

// ---------------------------------------------------------------------------
extern "C" void kernel_launch(void* const* d_in, const int* in_sizes, int n_in,
                              void* d_out, int out_size, void* d_ws, size_t ws_size,
                              hipStream_t stream)
{
    const float* x  = (const float*)d_in[0];
    const float* Wq = (const float*)d_in[1];
    const float* bq = (const float*)d_in[2];
    const float* Wk = (const float*)d_in[3];
    const float* bk = (const float*)d_in[4];
    const float* Wv = (const float*)d_in[5];
    const float* bv = (const float*)d_in[6];
    float* out = (float*)d_out;

    // workspace layout (bf16): xb | Wt[3] | Qb | Kb | VT   (~69 MB total)
    const size_t n  = (size_t)BATCH * SEQ * DHID;   // 8.39M elements
    const size_t wn = (size_t)DHID * DHID;          // 262K elements
    bf16_t* xb  = (bf16_t*)d_ws;
    bf16_t* WtB = xb + n;
    bf16_t* Qb  = WtB + 3 * wn;
    bf16_t* Kb  = Qb + n;
    bf16_t* VT  = Kb + n;

    cvt_x_kernel<<<dim3((unsigned)(n / (256 * 8))), 256, 0, stream>>>(x, xb);
    cvt_w_kernel<<<dim3(128, 1, 3), 256, 0, stream>>>(Wq, Wk, Wv, WtB);

    dim3 g1(BATCH * SEQ / 64, DHID / 64, 3);
    qkv_kernel<<<g1, 256, 0, stream>>>(xb, WtB, bq, bk, bv, Qb, Kb, VT);

    dim3 g2(SEQ / 32, BATCH, 1);
    attn_kernel<<<g2, 256, 0, stream>>>(Qb, Kb, VT, out);
}
